// TensorFusion_77704548319486
// MI455X (gfx1250) — matrix operand order repaired and checked by reading, not compiled, not run on hardware
//
#include <hip/hip_runtime.h>
#include <math.h>

typedef __attribute__((ext_vector_type(16))) __bf16 v16bf;
typedef __attribute__((ext_vector_type(8)))  float  v8f;
typedef __attribute__((ext_vector_type(4)))  unsigned int uint32x4;
typedef __attribute__((ext_vector_type(8)))  int          int32x8;
typedef __attribute__((ext_vector_type(4)))  int          int32x4;

#define D_L   64
#define DH    65
#define HID   64
#define BATCH 512
#define CUBE  (DH * DH * DH)

#define WMMA_BF16(A, B, C) \
  __builtin_amdgcn_wmma_f32_16x16x32_bf16(false, (A), false, (B), (short)0, (C), false, false)

// ---------------------------------------------------------------------------
// TDM: DMA one W1(i,j) slab (65 rows x 64 f32, contiguous) into LDS.
// D# built per ISA 8.3/8.4: 2D tensor, data_size=4B, tensor_dim0=64,
// tensor_dim1=65, tile 64x65, dim0_stride=64, type=2 ("image").
// This toolchain exposes the 6-arg builtin: (g0, g1, g2, g3, g_extra, cpol).
// ---------------------------------------------------------------------------
__device__ __forceinline__ void tdm_issue_slab(const float* W1, int i, int j,
                                               unsigned lds_off) {
  unsigned long long ga = (unsigned long long)(uintptr_t)W1 +
      ((unsigned long long)(i * DH + j) * DH) * (unsigned long long)(HID * 4);
  uint32x4 g0;
  g0[0] = 1u;                                   // count=1, user mode
  g0[1] = lds_off;                              // LDS byte address
  g0[2] = (unsigned)ga;                         // global_addr[31:0]
  g0[3] = (unsigned)(ga >> 32) | 0x80000000u;   // global_addr[56:32] | type=2
  int32x8 g1;
  g1[0] = 0x00020000;        // workgroup_mask=0, data_size=2 (4 bytes)
  g1[1] = (HID << 16);       // tensor_dim0[15:0]=64 at bits [31:16]
  g1[2] = (DH  << 16);       // tensor_dim0 hi=0; tensor_dim1[15:0]=65
  g1[3] = (HID << 16);       // tensor_dim1 hi=0; tile_dim0=64
  g1[4] = DH;                // tile_dim1=65; tile_dim2=0
  g1[5] = HID;               // tensor_dim0_stride=64 (low 32)
  g1[6] = 0;                 // stride hi / dim1_stride (unused, 2D)
  g1[7] = 0;
  int32x4 z4 = {0, 0, 0, 0};
  int32x8 z8 = {0, 0, 0, 0, 0, 0, 0, 0};
  __builtin_amdgcn_tensor_load_to_lds(g0, g1, z4, z4, z8, 0);
}

// ---------------------------------------------------------------------------
// Kernel 1: P[i, b, h] = a_h[b,i] * sum_j v_h[b,j] * (t_h @ W1[i,j,:,:])[b,h]
// Block = (64-row batch tile, i); 4 waves, each owns 16 h-columns and all
// 64 batch rows (4 sub-tiles sharing one B-fragment build -> 8 WMMA per j).
// W1 slabs stream through a TDM double buffer in LDS.
// ---------------------------------------------------------------------------
__global__ __launch_bounds__(128) void tf_fused_gemm(
    const float* __restrict__ l, const float* __restrict__ a,
    const float* __restrict__ v, const float* __restrict__ W1,
    float* __restrict__ P)
{
  __shared__ float  slab[2][DH * HID];   // 2 x 16.6 KB TDM staging
  __shared__ __bf16 t_s[64 * 64];        // t_h[b,k], k=0..63 (k=0 -> 1.0)
  __shared__ float  t64_s[64];           // t_h[b,64] = l[b,63]
  __shared__ float  v_s[64 * DH];        // v_h[b,j]
  __shared__ float  a_s[64];             // a_h[b,i]

  const int bt  = blockIdx.x;            // 0..7
  const int i   = blockIdx.y;            // 0..64
  const int b0  = bt * 64;
  const int tid = threadIdx.x;
  const int wave = tid >> 5;
  const int lane = tid & 31;
  const int lh   = lane >> 4;
  const int ln15 = lane & 15;
  const int h0   = wave * 16;

  // Kick off DMA of the first slab immediately (overlaps with preloads).
  if (wave == 0)
    tdm_issue_slab(W1, i, 0, (unsigned)(uintptr_t)&slab[0][0]);

  for (int idx = tid; idx < 64 * 64; idx += 128) {
    int row = idx >> 6, k = idx & 63;
    t_s[idx] = (__bf16)((k == 0) ? 1.0f : l[(b0 + row) * D_L + (k - 1)]);
  }
  for (int idx = tid; idx < 64 * DH; idx += 128) {
    int row = idx / DH, k = idx % DH;
    v_s[idx] = (k == 0) ? 1.0f : v[(b0 + row) * D_L + (k - 1)];
  }
  if (tid < 64) {
    t64_s[tid] = l[(b0 + tid) * D_L + 63];
    a_s[tid]   = (i == 0) ? 1.0f : a[(b0 + tid) * D_L + (i - 1)];
  }
  __syncthreads();

  // A fragments for the 4 row sub-tiles (16x32 bf16 layout, ISA 7.12.2);
  // j-invariant, register resident.
  v16bf A0[4], A1[4];
  float t64r[4][8];
#pragma unroll
  for (int st = 0; st < 4; ++st) {
    int rb = st * 16 + ln15;
#pragma unroll
    for (int e = 0; e < 16; ++e) {
      int kloc = (e < 8) ? (e + 8 * lh) : (8 + e + 8 * lh);
      A0[st][e] = t_s[rb * 64 + kloc];
      A1[st][e] = t_s[rb * 64 + 32 + kloc];
    }
#pragma unroll
    for (int r = 0; r < 8; ++r)
      t64r[st][r] = t64_s[st * 16 + lh * 8 + r];
  }

  v8f acc[4] = {};

  for (int j = 0; j < DH; ++j) {
    const int cur = j & 1;
    if (wave == 0)
      __builtin_amdgcn_s_wait_tensorcnt(0);   // slab[cur] landed
    __syncthreads();                          // visible to all; prev readers done
    if (wave == 0 && (j + 1) < DH)
      tdm_issue_slab(W1, i, j + 1, (unsigned)(uintptr_t)&slab[cur ^ 1][0]);

    const float* S = &slab[cur][0];

    // B fragments (32x16 bf16: lane = column, K = table order matching A), shared by
    // all 4 row sub-tiles.
    v16bf B0, B1;
#pragma unroll
    for (int e = 0; e < 16; ++e) {
      int k = (e < 8) ? (e + 8 * lh) : (8 + e + 8 * lh);
      B0[e] = (__bf16)S[k * HID + h0 + ln15];
      B1[e] = (__bf16)S[(k + 32) * HID + h0 + ln15];
    }
    float w64 = S[64 * HID + h0 + ln15];      // k = 64 remainder row

#pragma unroll
    for (int st = 0; st < 4; ++st) {
      v8f c = {};
      c = WMMA_BF16(A0[st], B0, c);
      c = WMMA_BF16(A1[st], B1, c);
#pragma unroll
      for (int r = 0; r < 8; ++r) {
        float vv = v_s[(st * 16 + lh * 8 + r) * DH + j];
        acc[st][r] += vv * (c[r] + t64r[st][r] * w64);
      }
    }
  }

  // Scale by a_h[b,i] and emit partial P[i, b, h]
  float* Pp = P + ((long long)i * BATCH + b0) * HID + h0 + ln15;
#pragma unroll
  for (int st = 0; st < 4; ++st)
#pragma unroll
    for (int r = 0; r < 8; ++r) {
      int row = st * 16 + lh * 8 + r;
      Pp[row * HID] = acc[st][r] * a_s[row];
    }
}

// ---------------------------------------------------------------------------
// Kernel 2: h1 = tanh(b1 + sum_i P[i, :, :])
// ---------------------------------------------------------------------------
__global__ __launch_bounds__(256) void tf_reduce(
    const float* __restrict__ P, const float* __restrict__ b1,
    float* __restrict__ h1)
{
  int idx = blockIdx.x * blockDim.x + threadIdx.x;
  if (idx >= BATCH * HID) return;
  float s = b1[idx & (HID - 1)];
#pragma unroll 5
  for (int i = 0; i < DH; ++i)
    s += P[(long long)i * (BATCH * HID) + idx];
  h1[idx] = tanhf(s);
}

// ---------------------------------------------------------------------------
// Kernel 3: out = tanh(tanh(h1@W2 + b2)@W3 + b3), WMMA per 16-row batch tile.
// ---------------------------------------------------------------------------
__global__ __launch_bounds__(128) void tf_mlp(
    const float* __restrict__ h1,
    const float* __restrict__ W2, const float* __restrict__ b2,
    const float* __restrict__ W3, const float* __restrict__ b3,
    float* __restrict__ out)
{
  __shared__ __bf16 hbuf[16 * 64];
  const int b0  = blockIdx.x * 16;
  const int tid = threadIdx.x;

  for (int idx = tid; idx < 16 * 64; idx += 128)
    hbuf[idx] = (__bf16)h1[(b0 + (idx >> 6)) * HID + (idx & 63)];
  __syncthreads();

  const int wave = tid >> 5;
  const int lane = tid & 31;
  const int lh   = lane >> 4;
  const int ln15 = lane & 15;
  const int h0   = wave * 16;
  const int mrow = lh * 8;

  // ---- layer 2: h2 = tanh(h1 @ W2 + b2) ----
  v16bf A0, A1, B0, B1;
#pragma unroll
  for (int e = 0; e < 16; ++e) {
    int kloc = (e < 8) ? (e + 8 * lh) : (8 + e + 8 * lh);
    int k    = kloc;
    A0[e] = hbuf[ln15 * 64 + kloc];
    A1[e] = hbuf[ln15 * 64 + 32 + kloc];
    B0[e] = (__bf16)W2[k * HID + h0 + ln15];
    B1[e] = (__bf16)W2[(k + 32) * HID + h0 + ln15];
  }
  v8f c = {};
  c = WMMA_BF16(A0, B0, c);
  c = WMMA_BF16(A1, B1, c);

  float res[8];
  float bias2 = b2[h0 + ln15];
#pragma unroll
  for (int r = 0; r < 8; ++r) res[r] = tanhf(c[r] + bias2);

  __syncthreads();                      // all reads of h1 tile done
#pragma unroll
  for (int r = 0; r < 8; ++r)
    hbuf[(mrow + r) * 64 + h0 + ln15] = (__bf16)res[r];
  __syncthreads();

  // ---- layer 3: out = tanh(h2 @ W3 + b3) ----
#pragma unroll
  for (int e = 0; e < 16; ++e) {
    int kloc = (e < 8) ? (e + 8 * lh) : (8 + e + 8 * lh);
    int k    = kloc;
    A0[e] = hbuf[ln15 * 64 + kloc];
    A1[e] = hbuf[ln15 * 64 + 32 + kloc];
    B0[e] = (__bf16)W3[k * HID + h0 + ln15];
    B1[e] = (__bf16)W3[(k + 32) * HID + h0 + ln15];
  }
  v8f c2 = {};
  c2 = WMMA_BF16(A0, B0, c2);
  c2 = WMMA_BF16(A1, B1, c2);

  float bias3 = b3[h0 + ln15];
#pragma unroll
  for (int r = 0; r < 8; ++r)
    out[(b0 + mrow + r) * HID + h0 + ln15] = tanhf(c2[r] + bias3);
}

// ---------------------------------------------------------------------------
extern "C" void kernel_launch(void* const* d_in, const int* in_sizes, int n_in,
                              void* d_out, int out_size, void* d_ws, size_t ws_size,
                              hipStream_t stream) {
  (void)in_sizes; (void)n_in; (void)out_size; (void)ws_size;
  const float* l  = (const float*)d_in[0];
  const float* a  = (const float*)d_in[1];
  const float* v  = (const float*)d_in[2];
  const float* W1 = (const float*)d_in[3];
  const float* b1 = (const float*)d_in[4];
  const float* W2 = (const float*)d_in[5];
  const float* b2 = (const float*)d_in[6];
  const float* W3 = (const float*)d_in[7];
  const float* b3 = (const float*)d_in[8];
  float* out = (float*)d_out;

  float* P  = (float*)d_ws;                       // [65, 512, 64] f32 partials
  float* h1 = P + (size_t)DH * BATCH * HID;       // [512, 64]

  tf_fused_gemm<<<dim3(8, DH), 128, 0, stream>>>(l, a, v, W1, P);
  tf_reduce<<<(BATCH * HID + 255) / 256, 256, 0, stream>>>(P, b1, h1);
  tf_mlp<<<BATCH / 16, 128, 0, stream>>>(h1, W2, b2, W3, b3, out);
}
